// Attention_55508157334193
// MI455X (gfx1250) — compile-verified, matrix operand order checked by reading, not run on hardware
//
#include <hip/hip_runtime.h>
#include <hip/hip_bf16.h>

// Problem constants (match setup_inputs): B=2, N=2048, D=2048, H=16, hd=128.
#define B_  2
#define N_  2048
#define D_  2048
#define H_  16
#define HD_ 128
#define K3_ (3 * D_)   // 6144

typedef __attribute__((ext_vector_type(16))) __bf16 v16bf;
typedef __attribute__((ext_vector_type(8)))  __bf16 v8bf;
typedef __attribute__((ext_vector_type(4)))  __bf16 v4bf;
typedef __attribute__((ext_vector_type(8)))  float  v8f;
typedef __attribute__((ext_vector_type(4)))  float  v4f;

// ---------------------------------------------------------------------------
// WMMA helper: D = A(16x32 bf16) * B(32x16 bf16) + C(16x16 f32)
// ---------------------------------------------------------------------------
__device__ __forceinline__ v8f wmma_bf16(v16bf a, v16bf b, v8f c) {
  return __builtin_amdgcn_wmma_f32_16x16x32_bf16(
      /*neg_a=*/false, a, /*neg_b=*/false, b,
      /*c_mod=*/(short)0, c, /*reuse_a=*/false, /*reuse_b=*/false);
}

// ---------------------------------------------------------------------------
// Fragment loader (bf16 source). ISA 7.12.2 (16-bit A/B 16x32): lane holds
// row = lane%16, half = lane/16; element j<8 -> K = k0 + 8*half + j,
//                                element j>=8 -> K = k0 + 16 + 8*half + (j-8).
// Works for global or LDS source (two contiguous 16B groups per lane).
// ---------------------------------------------------------------------------
__device__ __forceinline__ v16bf load_frag_bf16(const __bf16* __restrict__ row,
                                                int k0, int hf) {
  v8bf a = *reinterpret_cast<const v8bf*>(row + k0 + 8 * hf);
  v8bf b = *reinterpret_cast<const v8bf*>(row + k0 + 16 + 8 * hf);
  v16bf f;
#pragma unroll
  for (int j = 0; j < 8; ++j) {
    f[j]     = a[j];
    f[8 + j] = b[j];
  }
  return f;
}

// ---------------------------------------------------------------------------
// Async staging of a 128x32 bf16 B tile (rows = output cols, contiguous K)
// into LDS via the CDNA5 async copy path (ASYNCcnt-tracked).
// 128 rows * 64B = 512 chunks of 16B; 2 chunks per thread (256 threads).
// LDS offset = low 32 bits of the generic pointer (LDS aperture, ISA 10.2).
// ---------------------------------------------------------------------------
__device__ __forceinline__ void stage_b_async(const __bf16* __restrict__ W,
                                              int c0, int k0,
                                              __bf16* buf, int tid) {
#pragma unroll
  for (int i = 0; i < 2; ++i) {
    int task = tid * 2 + i;       // 0..511
    int row  = task >> 2;         // 0..127
    int ch   = task & 3;          // 16B chunk within 64B row
    const __bf16* g = W + (size_t)(c0 + row) * D_ + k0 + ch * 8;
    unsigned int  lds = (unsigned int)(uintptr_t)(buf + row * 32 + ch * 8);
    unsigned long long ga = (unsigned long long)(uintptr_t)g;
    asm volatile("global_load_async_to_lds_b128 %0, %1, off"
                 :: "v"(lds), "v"(ga) : "memory");
  }
}

__device__ __forceinline__ void wait_async0() {
  asm volatile("s_wait_asynccnt 0x0" ::: "memory");
}

// ---------------------------------------------------------------------------
// Kernel 0: fp32 -> bf16 conversion (4 elements / thread, n % 4 == 0)
// ---------------------------------------------------------------------------
__global__ __launch_bounds__(256) void cvt_bf16_kernel(
    const float* __restrict__ in, __bf16* __restrict__ outp, int n) {
  int base = (blockIdx.x * 256 + threadIdx.x) * 4;
  if (base + 3 >= n) return;
  v4f v = *reinterpret_cast<const v4f*>(in + base);
  v4bf o;
#pragma unroll
  for (int j = 0; j < 4; ++j) o[j] = (__bf16)v[j];
  *reinterpret_cast<v4bf*>(outp + base) = o;
}

// ---------------------------------------------------------------------------
// Shared GEMM mainloop: block computes a 128x128 tile of C = A * W^T.
// 8 waves; wave w owns rows [m0, m0+16). B tile double-buffered in LDS via
// async copies; each wave runs 8 WMMAs per 32-wide K step.
// Returns accumulators in acc[8] (d-tile t covers cols c0+16t .. c0+16t+15).
// ---------------------------------------------------------------------------
__device__ __forceinline__ void gemm_mainloop_128(
    const __bf16* __restrict__ A, const __bf16* __restrict__ W,
    int m_row,                   // this lane's A row (m0 + lane%16)
    int c0,                      // block column base
    __bf16 (*bt)[128][32],       // LDS double buffer
    int tid, int hf, int lr, v8f acc[8]) {
  const __bf16* arow = A + (size_t)m_row * D_;

  stage_b_async(W, c0, 0, &bt[0][0][0], tid);
  wait_async0();
  __syncthreads();

  int buf = 0;
  for (int k0 = 0; k0 < D_; k0 += 32) {
    if (k0 + 32 < D_)
      stage_b_async(W, c0, k0 + 32, &bt[buf ^ 1][0][0], tid);

    __builtin_prefetch(arow + k0 + 256, 0, 0);   // global_prefetch_b8
    v16bf a = load_frag_bf16(arow, k0, hf);
#pragma unroll
    for (int t = 0; t < 8; ++t) {
      v16bf b = load_frag_bf16(&bt[buf][16 * t + lr][0], 0, hf);
      acc[t] = wmma_bf16(a, b, acc[t]);
    }
    wait_async0();
    __syncthreads();
    buf ^= 1;
  }
}

// ---------------------------------------------------------------------------
// Kernel 1: qkv[m,e] = sum_d x[m,d] * Wqkv[e,d]   (bf16 in, bf16 out)
// ---------------------------------------------------------------------------
__global__ __launch_bounds__(256) void qkv_gemm_kernel(
    const __bf16* __restrict__ X, const __bf16* __restrict__ W,
    __bf16* __restrict__ C) {
  __shared__ __bf16 bt[2][128][32];   // 16 KB double buffer
  const int tid  = threadIdx.x;
  const int lane = tid & 31, wave = tid >> 5;
  const int nblk = K3_ / 128;         // 48
  const int mb = blockIdx.x / nblk;
  const int nb = blockIdx.x % nblk;
  const int m0 = mb * 128 + wave * 16;
  const int c0 = nb * 128;
  const int hf = lane >> 4, lr = lane & 15;

  v8f acc[8];
#pragma unroll
  for (int t = 0; t < 8; ++t)
#pragma unroll
    for (int r = 0; r < 8; ++r) acc[t][r] = 0.0f;

  gemm_mainloop_128(X, W, m0 + lr, c0, bt, tid, hf, lr, acc);

  // C/D layout: col = lane%16, row = r + 8*half.
#pragma unroll
  for (int t = 0; t < 8; ++t)
#pragma unroll
    for (int r = 0; r < 8; ++r) {
      int row = m0 + r + 8 * hf;
      int col = c0 + 16 * t + lr;
      C[(size_t)row * K3_ + col] = (__bf16)acc[t][r];
    }
}

// ---------------------------------------------------------------------------
// Kernel 2: RoPE on q and k halves of qkv (in place, bf16).
// ---------------------------------------------------------------------------
__global__ __launch_bounds__(256) void rope_kernel(__bf16* __restrict__ qkv) {
  int idx = blockIdx.x * blockDim.x + threadIdx.x;
  const int total = B_ * N_ * H_ * (HD_ / 2);
  if (idx >= total) return;
  int i = idx % (HD_ / 2);
  int t = idx / (HD_ / 2);
  int h = t % H_;  t /= H_;
  int n = t % N_;
  int b = t / N_;
  float freq = __powf(10000.0f, -(float)(2 * i) / (float)HD_);
  float ang  = (float)n * freq;
  float s, c;
  __sincosf(ang, &s, &c);
  size_t base = ((size_t)b * N_ + n) * K3_ + (size_t)h * HD_ + i;
  float q1 = (float)qkv[base], q2 = (float)qkv[base + HD_ / 2];
  qkv[base]           = (__bf16)(q1 * c - q2 * s);
  qkv[base + HD_ / 2] = (__bf16)(q1 * s + q2 * c);
  size_t kb = base + D_;
  float k1 = (float)qkv[kb], k2 = (float)qkv[kb + HD_ / 2];
  qkv[kb]             = (__bf16)(k1 * c - k2 * s);
  qkv[kb + HD_ / 2]   = (__bf16)(k1 * s + k2 * c);
}

// ---------------------------------------------------------------------------
// Kernel 3: flash attention, one wave per (b, h, 16-query tile).
// S^T = K * Q^T: each lane owns one query column; softmax reductions are
// in-lane + one shfl_xor(16); P^T B-fragment packs purely in-lane.
// O^T accumulated as 8 WMMA tiles: O^T += V^T * P^T (V^T staged via LDS).
// ---------------------------------------------------------------------------
__global__ __launch_bounds__(128) void attn_kernel(
    const __bf16* __restrict__ qkv, const int* __restrict__ L,
    __bf16* __restrict__ AO) {
  __shared__ __bf16 vt[4][HD_][32];          // V^T staging: [d][key] per wave

  const int lane = threadIdx.x & 31;
  const int w    = threadIdx.x >> 5;
  const int gw   = blockIdx.x * 4 + w;       // 4096 waves total
  const int qt = gw & (N_ / 16 - 1);         // 0..127
  const int h  = (gw >> 7) & (H_ - 1);
  const int b  = gw >> 11;
  const int q0 = qt * 16;
  const int hf = lane >> 4, lr = lane & 15;
  const int Lb = L[b];
  const float scale = 0.08838834764831845f;  // 1/sqrt(128)

  const __bf16* qbase = qkv + (size_t)b * N_ * K3_ + (size_t)h * HD_;
  const __bf16* kbase = qbase + D_;
  const __bf16* vbase = qbase + 2 * D_;

  // Q^T B-fragments (constant over the key loop): B[kk=d][n=q] = Q[q, d]
  v16bf qf[4];
  {
    const __bf16* qrow = qbase + (size_t)(q0 + lr) * K3_;
#pragma unroll
    for (int c = 0; c < 4; ++c) qf[c] = load_frag_bf16(qrow, 32 * c, hf);
  }

  v8f o[8];
#pragma unroll
  for (int t = 0; t < 8; ++t)
#pragma unroll
    for (int r = 0; r < 8; ++r) o[t][r] = 0.0f;

  float m = -1.0e30f, l = 0.0f;
  const int q    = q0 + lr;                  // this lane's query
  const int kmax = q0 + 16;                  // causal upper bound (exclusive)

  for (int k0 = 0; k0 < kmax; k0 += 32) {
    // ---- scores^T: two 16-key tiles, each = sum of 4 WMMAs over d ----
    v8f s0, s1;
#pragma unroll
    for (int r = 0; r < 8; ++r) { s0[r] = 0.0f; s1[r] = 0.0f; }
#pragma unroll
    for (int c = 0; c < 4; ++c) {
      const __bf16* krow0 = kbase + (size_t)(k0 + lr) * K3_;
      v16bf a0 = load_frag_bf16(krow0, 32 * c, hf);
      s0 = wmma_bf16(a0, qf[c], s0);
    }
#pragma unroll
    for (int c = 0; c < 4; ++c) {
      const __bf16* krow1 = kbase + (size_t)(k0 + 16 + lr) * K3_;
      v16bf a1 = load_frag_bf16(krow1, 32 * c, hf);
      s1 = wmma_bf16(a1, qf[c], s1);
    }

    // ---- scale + causal/length mask, per-query running max ----
    float tmax = -1.0e30f;
#pragma unroll
    for (int r = 0; r < 8; ++r) {
      int key0 = k0 + r + 8 * hf;
      int key1 = key0 + 16;
      float a  = s0[r] * scale;
      float c2 = s1[r] * scale;
      a  = (q >= key0 && key0 < Lb) ? a  : -1.0e9f;
      c2 = (q >= key1 && key1 < Lb) ? c2 : -1.0e9f;
      s0[r] = a; s1[r] = c2;
      tmax = fmaxf(tmax, fmaxf(a, c2));
    }
    tmax = fmaxf(tmax, __shfl_xor(tmax, 16, 32));
    float mnew  = fmaxf(m, tmax);
    float alpha = __expf(m - mnew);

    // ---- exp + per-query partial sum; pack P^T B-fragment in-lane ----
    float psum = 0.0f;
    v16bf pb;
#pragma unroll
    for (int r = 0; r < 8; ++r) {
      float p0 = __expf(s0[r] - mnew);
      float p1 = __expf(s1[r] - mnew);
      psum += p0 + p1;
      pb[r]     = (__bf16)p0;   // kk = 8*half + r      (keys k0 .. k0+15)
      pb[8 + r] = (__bf16)p1;   // kk = 16 + 8*half + r (keys k0+16 .. k0+31)
    }
    psum += __shfl_xor(psum, 16, 32);
    l = l * alpha + psum;
    m = mnew;

#pragma unroll
    for (int t = 0; t < 8; ++t)
#pragma unroll
      for (int r = 0; r < 8; ++r) o[t][r] *= alpha;

    // ---- stage V^T into LDS: lane owns key k0+lane, coalesced row read ----
    {
      int key = k0 + lane;
      if (key < N_) {
        const __bf16* vrow = vbase + (size_t)key * K3_;
#pragma unroll 8
        for (int d = 0; d < HD_; ++d) vt[w][d][lane] = vrow[d];
      } else {
#pragma unroll 8
        for (int d = 0; d < HD_; ++d) vt[w][d][lane] = (__bf16)0.0f;
      }
    }
    // cross-lane LDS dependency within the wave: drain DS counter
    asm volatile("s_wait_dscnt 0" ::: "memory");

    // ---- O^T += V^T(16d x 32k) * P^T(32k x 16q), 8 d-tiles ----
#pragma unroll
    for (int t = 0; t < 8; ++t) {
      const __bf16* vr = &vt[w][16 * t + lr][0];
      v16bf a;
#pragma unroll
      for (int j = 0; j < 8; ++j) {
        a[j]     = vr[8 * hf + j];
        a[8 + j] = vr[16 + 8 * hf + j];
      }
      o[t] = wmma_bf16(a, pb, o[t]);
    }
  }

  // ---- normalize + write AO[b, q, h*128 + d] (bf16) ----
  float invl = 1.0f / l;
  __bf16* aorow = AO + ((size_t)b * N_ + q) * D_ + (size_t)h * HD_;
#pragma unroll
  for (int t = 0; t < 8; ++t)
#pragma unroll
    for (int r = 0; r < 8; ++r)
      aorow[16 * t + r + 8 * hf] = (__bf16)(o[t][r] * invl);
}

// ---------------------------------------------------------------------------
// Kernel 4: out[m,e] = sum_d AO[m,d] * Wproj[e,d] + bproj[e]   (fp32 out)
// ---------------------------------------------------------------------------
__global__ __launch_bounds__(256) void proj_gemm_kernel(
    const __bf16* __restrict__ A, const __bf16* __restrict__ W,
    const float* __restrict__ bias, float* __restrict__ out) {
  __shared__ __bf16 bt[2][128][32];
  const int tid  = threadIdx.x;
  const int lane = tid & 31, wave = tid >> 5;
  const int nblk = D_ / 128;          // 16
  const int mb = blockIdx.x / nblk;
  const int nb = blockIdx.x % nblk;
  const int m0 = mb * 128 + wave * 16;
  const int c0 = nb * 128;
  const int hf = lane >> 4, lr = lane & 15;

  v8f acc[8];
#pragma unroll
  for (int t = 0; t < 8; ++t)
#pragma unroll
    for (int r = 0; r < 8; ++r) acc[t][r] = 0.0f;

  gemm_mainloop_128(A, W, m0 + lr, c0, bt, tid, hf, lr, acc);

#pragma unroll
  for (int t = 0; t < 8; ++t)
#pragma unroll
    for (int r = 0; r < 8; ++r) {
      int row = m0 + r + 8 * hf;
      int col = c0 + 16 * t + lr;
      out[(size_t)row * D_ + col] = acc[t][r] + bias[col];
    }
}

// ---------------------------------------------------------------------------
// Launch: x, Wqkv, Wproj, bproj, L, n_heads  ->  out (fp32)
// Workspace layout (bytes):
//   [0)            qkv bf16   B*N*3D          = 50331648
//   [50331648)     AO  bf16   B*N*D           = 16777216
//   [67108864)     x   bf16   B*N*D           = 16777216
//   [83886080)     Wqkv bf16  3D*D            = 25165824
//   [109051904)    Wproj bf16 D*D             =  8388608   (end: 117440512)
// ---------------------------------------------------------------------------
extern "C" void kernel_launch(void* const* d_in, const int* in_sizes, int n_in,
                              void* d_out, int out_size, void* d_ws,
                              size_t ws_size, hipStream_t stream) {
  const float* x     = (const float*)d_in[0];
  const float* Wqkv  = (const float*)d_in[1];
  const float* Wproj = (const float*)d_in[2];
  const float* bproj = (const float*)d_in[3];
  const int*   L     = (const int*)d_in[4];
  float*       out   = (float*)d_out;

  char* ws = (char*)d_ws;
  __bf16* qkv      = (__bf16*)(ws);
  __bf16* ao       = (__bf16*)(ws + 50331648ull);
  __bf16* x_bf     = (__bf16*)(ws + 67108864ull);
  __bf16* wqkv_bf  = (__bf16*)(ws + 83886080ull);
  __bf16* wproj_bf = (__bf16*)(ws + 109051904ull);

  const int n_x  = B_ * N_ * D_;     // 8388608
  const int n_wq = K3_ * D_;         // 12582912
  const int n_wp = D_ * D_;          // 4194304

  cvt_bf16_kernel<<<n_x  / 1024, 256, 0, stream>>>(x,     x_bf,     n_x);
  cvt_bf16_kernel<<<n_wq / 1024, 256, 0, stream>>>(Wqkv,  wqkv_bf,  n_wq);
  cvt_bf16_kernel<<<n_wp / 1024, 256, 0, stream>>>(Wproj, wproj_bf, n_wp);

  // QKV GEMM: (4096/128) x (6144/128) = 32*48 = 1536 blocks of 8 waves
  qkv_gemm_kernel<<<1536, 256, 0, stream>>>(x_bf, wqkv_bf, qkv);

  // RoPE: B*N*H*64 threads
  const int rope_total = B_ * N_ * H_ * (HD_ / 2);
  rope_kernel<<<(rope_total + 255) / 256, 256, 0, stream>>>(qkv);

  // Attention: B*H*(N/16) = 4096 waves -> 1024 blocks of 4 waves (32KB LDS)
  attn_kernel<<<1024, 128, 0, stream>>>(qkv, L, ao);

  // Projection: (4096/128) x (2048/128) = 32*16 = 512 blocks of 8 waves
  proj_gemm_kernel<<<512, 256, 0, stream>>>(ao, wproj_bf, bproj, out);
}
